// MultiDimensionalGCN_6_split_27565100105915
// MI455X (gfx1250) — matrix operand order repaired and checked by reading, not compiled, not run on hardware
//
#include <hip/hip_runtime.h>
#include <hip/hip_bf16.h>

// ---------------------------------------------------------------------------
// MultiDimensionalGCN forward for MI455X (gfx1250, wave32, WMMA).
// GEMM pipeline: operands converted to bf16 once (weights transposed to
// [N,K]), tiles staged global->LDS with GLOBAL_LOAD_ASYNC_TO_LDS_B128
// (ASYNCcnt + barrier), fragments read as ds_load_b128 per the ISA VGPR
// layouts, 4x v_wmma_f32_16x16x32_bf16 per wave per K-step (A reuse).
// Eigenvalues via parallel-ordered cyclic Jacobi (single 1024-thread WG).
// ---------------------------------------------------------------------------

#define NNODES 2048
#define NEDGES 65536
#define NDIMS  5
#define CMAX   256
#define THRESH 0.5f
#define JACOBI_SWEEPS 2

typedef __attribute__((ext_vector_type(16))) __bf16 v16bf;
typedef __attribute__((ext_vector_type(8)))  __bf16 bf16x8;
typedef __attribute__((ext_vector_type(8)))  float  v8f;

static __device__ __forceinline__ int gidx() {
    return blockIdx.x * blockDim.x + threadIdx.x;
}

// ------------------------------ elementwise --------------------------------

__global__ void k_zero(float* __restrict__ p, int n) {
    int i = gidx();
    if (i < n) p[i] = 0.0f;
}

__global__ void k_cvt(const float* __restrict__ in, __bf16* __restrict__ out, int n) {
    int i = gidx();
    if (i < n) out[i] = (__bf16)in[i];
}

// W[K,N] f32 -> out[N,K] bf16 (transpose during convert; weights are small)
__global__ void k_cvt_t(const float* __restrict__ in, __bf16* __restrict__ out,
                        int K, int N) {
    int idx = gidx();
    if (idx >= K * N) return;
    int k = idx / N, n = idx - k * N;
    out[(size_t)n * K + k] = (__bf16)in[idx];
}

__global__ void k_deg_scatter(const int* __restrict__ dst,
                              const float* __restrict__ ew,
                              float* __restrict__ deg, int E) {
    int e = gidx();
    if (e < E) atomicAdd(&deg[dst[e]], ew[e]);
}

__global__ void k_deg_finish(float* __restrict__ deg, int n) {
    int i = gidx();
    if (i < n) deg[i] += 1.0f;
}

__global__ void k_coef(const int* __restrict__ src, const int* __restrict__ dst,
                       const float* __restrict__ ew, const float* __restrict__ deg,
                       float* __restrict__ coef, int E) {
    int e = gidx();
    if (e < E) coef[e] = ew[e] * rsqrtf(deg[src[e]]) * rsqrtf(deg[dst[e]]);
}

__global__ void k_aggregate(const int* __restrict__ src, const int* __restrict__ dst,
                            const float* __restrict__ coef,
                            const float* __restrict__ h,
                            float* __restrict__ agg, int E, int C) {
    int idx = gidx();
    int quads = C >> 2;
    if (idx >= E * quads) return;
    int e  = idx / quads;
    int cq = (idx - e * quads) << 2;
    float cf = coef[e];
    const float4 hv = *(const float4*)(h + (size_t)src[e] * C + cq);
    float* a = agg + (size_t)dst[e] * C + cq;
    atomicAdd(a + 0, hv.x * cf);
    atomicAdd(a + 1, hv.y * cf);
    atomicAdd(a + 2, hv.z * cf);
    atomicAdd(a + 3, hv.w * cf);
}

__global__ void k_gcn_finish(const float* __restrict__ agg, const float* __restrict__ h,
                             const float* __restrict__ deg, const float* __restrict__ b,
                             float* __restrict__ g, int C) {
    int idx = gidx();
    if (idx >= NNODES * C) return;
    int n = idx / C, c = idx - n * C;
    g[idx] = agg[idx] + h[idx] / deg[n] + b[c];
}

// row-normalize and emit bf16 directly (feeds the Gram WMMA)
__global__ void k_rownorm(const float* __restrict__ g, __bf16* __restrict__ xn, int C) {
    __shared__ float red[256];
    int n = blockIdx.x, t = threadIdx.x;
    float s = 0.0f;
    for (int c = t; c < C; c += 256) { float v = g[(size_t)n * C + c]; s += v * v; }
    red[t] = s; __syncthreads();
    for (int o = 128; o > 0; o >>= 1) { if (t < o) red[t] += red[t + o]; __syncthreads(); }
    float rinv = 1.0f / (sqrtf(red[0]) + 1e-8f);
    for (int c = t; c < C; c += 256) xn[(size_t)n * C + c] = (__bf16)(g[(size_t)n * C + c] * rinv);
}

__global__ void k_laplacian(float* __restrict__ S, int n) {
    __shared__ float red[256];
    int r = blockIdx.x, t = threadIdx.x;
    float s = 0.0f;
    for (int j = t; j < n; j += 256) s += S[(size_t)r * n + j];
    red[t] = s; __syncthreads();
    for (int o = 128; o > 0; o >>= 1) { if (t < o) red[t] += red[t + o]; __syncthreads(); }
    float rs = red[0]; __syncthreads();
    for (int j = t; j < n; j += 256) {
        float v = S[(size_t)r * n + j];
        S[(size_t)r * n + j] = (j == r) ? (rs - v) : (-v);
    }
}

// ------------------------------- WMMA GEMM ---------------------------------
// C[M,N] = A[M,K] @ Bt^T (+bias, optional relu). A bf16 [M,K]; Bt bf16 [N,K].
// Block = 256 threads = 8 waves; block tile 128(M) x 64(N); K-step 32.
// Tiles staged with GLOBAL_LOAD_ASYNC_TO_LDS_B128 (16B chunks, GVS mode),
// completed with s_wait_asynccnt + workgroup barrier. Wave w computes
// rows [w*16,w*16+16) x 64 cols: one A fragment reused by 4 WMMAs.
__global__ __launch_bounds__(256)
void k_gemm_bf16(const __bf16* __restrict__ A, const __bf16* __restrict__ Bt,
                 const float* __restrict__ bias, float* __restrict__ C,
                 int M, int K, int N, int relu) {
    __shared__ __align__(16) __bf16 lA[128 * 32];   // [row][k] 8KB
    __shared__ __align__(16) __bf16 lB[64 * 32];    // [col][k] 4KB
    const int tid  = threadIdx.x;
    const int wave = tid >> 5;
    const int lane = tid & 31;
    const int lrow = lane & 15;
    const int hf   = lane >> 4;
    const int colBase = blockIdx.x * 64;
    const int rowBase = blockIdx.y * 128;

    const unsigned long long aBase = (unsigned long long)A;
    const unsigned long long bBase = (unsigned long long)Bt;
    // LDS byte addresses (low 32 bits of the generic LDS pointer)
    const unsigned lAoff = (unsigned)(unsigned long long)(const void*)lA;
    const unsigned lBoff = (unsigned)(unsigned long long)(const void*)lB;

    // Per-thread async chunk map: A = 512 x 16B chunks (2/thread), B = 256 (1/thread).
    // Row stride K*2 bytes (K multiple of 16 -> 16B aligned chunks).
    const int ca0 = tid, ca1 = tid + 256, cb = tid;
    unsigned lA0 = lAoff + (unsigned)(ca0 << 4);
    unsigned lA1 = lAoff + (unsigned)(ca1 << 4);
    unsigned lB0 = lBoff + (unsigned)(cb << 4);
    unsigned gA0 = (unsigned)(((rowBase + (ca0 >> 2)) * K) << 1) + ((ca0 & 3) << 4);
    unsigned gA1 = (unsigned)(((rowBase + (ca1 >> 2)) * K) << 1) + ((ca1 & 3) << 4);
    unsigned gB0 = (unsigned)(((colBase + (cb  >> 2)) * K) << 1) + ((cb  & 3) << 4);

    v8f acc[4];
#pragma unroll
    for (int j = 0; j < 4; ++j) acc[j] = (v8f){0.f,0.f,0.f,0.f,0.f,0.f,0.f,0.f};

    for (int k0 = 0; k0 < K; k0 += 32) {
        // ---- async stage: global bf16 tiles -> LDS, no VGPR data path ----
        asm volatile("global_load_async_to_lds_b128 %0, %1, %2"
                     :: "v"(lA0), "v"(gA0), "s"(aBase) : "memory");
        asm volatile("global_load_async_to_lds_b128 %0, %1, %2"
                     :: "v"(lA1), "v"(gA1), "s"(aBase) : "memory");
        asm volatile("global_load_async_to_lds_b128 %0, %1, %2"
                     :: "v"(lB0), "v"(gB0), "s"(bBase) : "memory");
        asm volatile("s_wait_asynccnt 0x0" ::: "memory");
        __syncthreads();

        // ---- A fragment: two contiguous 8x bf16 chunks per ISA layout ----
        const __bf16* ap = lA + (wave * 16 + lrow) * 32;
        bf16x8 a0 = *(const bf16x8*)(ap + 8 * hf);
        bf16x8 a1 = *(const bf16x8*)(ap + 16 + 8 * hf);
        v16bf af = __builtin_shufflevector(a0, a1, 0, 1, 2, 3, 4, 5, 6, 7,
                                           8, 9, 10, 11, 12, 13, 14, 15);
        // ---- 4 B fragments, 4 WMMAs, reuse af ----
#pragma unroll
        for (int j = 0; j < 4; ++j) {
            const __bf16* bp = lB + (j * 16 + lrow) * 32 + 8 * hf;
            bf16x8 b0 = *(const bf16x8*)(bp + 0);
            bf16x8 b1 = *(const bf16x8*)(bp + 16);
            v16bf bfv = __builtin_shufflevector(b0, b1, 0, 1, 2, 3, 4, 5, 6, 7,
                                                8, 9, 10, 11, 12, 13, 14, 15);
            acc[j] = __builtin_amdgcn_wmma_f32_16x16x32_bf16(
                false, af, false, bfv, (short)0, acc[j], false, false);
        }
        __syncthreads();

        gA0 += 64; gA1 += 64; gB0 += 64;   // K-step advance: 32 bf16 = 64B
    }

    // ---- epilogue: C/D VGPR layout -> global ----
#pragma unroll
    for (int j = 0; j < 4; ++j) {
        int col = colBase + j * 16 + lrow;
        float bb = bias ? bias[col] : 0.0f;
#pragma unroll
        for (int r = 0; r < 8; ++r) {
            int row = rowBase + wave * 16 + r + 8 * hf;
            float v = acc[j][r] + bb;
            if (relu) v = fmaxf(v, 0.0f);
            C[(size_t)row * N + col] = v;
        }
    }
}

// --------------------------- Jacobi eigensolver ----------------------------
__global__ __launch_bounds__(1024)
void k_jacobi(float* __restrict__ A, int n, int sweeps) {
    __shared__ int   PP[1024], QQ[1024];
    __shared__ float CS[1024], SN[1024];
    const int t  = threadIdx.x;
    const int m  = n - 1;
    const int np = n >> 1;
    for (int sw = 0; sw < sweeps; ++sw) {
        for (int r = 0; r < m; ++r) {
            if (t < np) {
                int a, b;
                if (t == 0) { a = n - 1; b = r % m; }
                else        { a = (r + t) % m; b = (r - t + m) % m; }
                int p = a < b ? a : b, q = a < b ? b : a;
                float app = A[(size_t)p * n + p];
                float aqq = A[(size_t)q * n + q];
                float apq = A[(size_t)p * n + q];
                float c = 1.0f, s = 0.0f;
                if (fabsf(apq) > 1e-12f) {
                    float th = (aqq - app) / (2.0f * apq);
                    float tt = (th >= 0.0f ? 1.0f : -1.0f) /
                               (fabsf(th) + sqrtf(1.0f + th * th));
                    c = rsqrtf(1.0f + tt * tt);
                    s = tt * c;
                }
                PP[t] = p; QQ[t] = q; CS[t] = c; SN[t] = s;
            }
            __syncthreads();
            for (int idx = t; idx < np * n; idx += 1024) {
                int i = idx / n, j = idx - i * n;
                int p = PP[i], q = QQ[i];
                float c = CS[i], s = SN[i];
                float apj = A[(size_t)p * n + j], aqj = A[(size_t)q * n + j];
                A[(size_t)p * n + j] = c * apj - s * aqj;
                A[(size_t)q * n + j] = s * apj + c * aqj;
            }
            __syncthreads();
            for (int idx = t; idx < np * n; idx += 1024) {
                int i = idx / n, j = idx - i * n;
                int p = PP[i], q = QQ[i];
                float c = CS[i], s = SN[i];
                float ajp = A[(size_t)j * n + p], ajq = A[(size_t)j * n + q];
                A[(size_t)j * n + p] = c * ajp - s * ajq;
                A[(size_t)j * n + q] = s * ajp + c * ajq;
            }
            __syncthreads();
        }
    }
}

// Diagonal -> bitonic sort (LDS) -> eigengap threshold -> cumulative class ids.
__global__ __launch_bounds__(1024)
void k_eigflags(const float* __restrict__ L, int n, float thr, int* __restrict__ flags) {
    __shared__ float ev[2048];
    const int t = threadIdx.x;
    for (int i = t; i < n; i += 1024) ev[i] = L[(size_t)i * n + i];
    __syncthreads();
    for (int k = 2; k <= n; k <<= 1) {
        for (int j = k >> 1; j > 0; j >>= 1) {
            for (int i = t; i < n; i += 1024) {
                int ixj = i ^ j;
                if (ixj > i) {
                    bool up = ((i & k) == 0);
                    float a = ev[i], b = ev[ixj];
                    if ((a > b) == up) { ev[i] = b; ev[ixj] = a; }
                }
            }
            __syncthreads();
        }
    }
    if (t == 0) {
        int acc = 0;
        flags[0] = 0;
        for (int i = 1; i < n; ++i) {
            if (ev[i] - ev[i - 1] > thr) ++acc;
            flags[i] = acc;
        }
    }
}

// ------------------------------ class average ------------------------------

__global__ void k_class_sums(const int* __restrict__ flags, const float* __restrict__ g,
                             float* __restrict__ csum, float* __restrict__ ccnt, int C) {
    int idx = gidx();
    if (idx >= NNODES * C) return;
    int n = idx / C, c = idx - n * C;
    int f = flags[n];
    atomicAdd(&csum[(size_t)f * C + c], g[idx]);
    if (c == 0) atomicAdd(&ccnt[f], 1.0f);
}

__global__ void k_class_gather(const int* __restrict__ flags, const float* __restrict__ csum,
                               const float* __restrict__ ccnt, float* __restrict__ avg, int C) {
    int idx = gidx();
    if (idx >= NNODES * C) return;
    int n = idx / C, c = idx - n * C;
    int f = flags[n];
    avg[idx] = csum[(size_t)f * C + c] / fmaxf(ccnt[f], 1.0f);
}

// out = relu(fir + (fir + k*lin)) = relu(2*fir + k*lin)
__global__ void k_combine(const float* __restrict__ fir, const float* __restrict__ lin,
                          const float* __restrict__ kptr, float* __restrict__ out, int total) {
    int idx = gidx();
    if (idx >= total) return;
    float k = *kptr;
    out[idx] = fmaxf(2.0f * fir[idx] + k * lin[idx], 0.0f);
}

__global__ void k_final(const float* __restrict__ o0, const float* __restrict__ o1,
                        const float* __restrict__ o2, const float* __restrict__ o3,
                        const float* __restrict__ o4, const float* __restrict__ dw,
                        float* __restrict__ out, int total) {
    int idx = gidx();
    if (idx >= total) return;
    float s = dw[0] + dw[1] + dw[2] + dw[3] + dw[4];
    out[idx] = (dw[0] * o0[idx] + dw[1] * o1[idx] + dw[2] * o2[idx] +
                dw[3] * o3[idx] + dw[4] * o4[idx]) / s;
}

// -------------------------------- host side --------------------------------

extern "C" void kernel_launch(void* const* d_in, const int* in_sizes, int n_in,
                              void* d_out, int out_size, void* d_ws, size_t ws_size,
                              hipStream_t stream) {
    (void)in_sizes; (void)n_in; (void)out_size; (void)ws_size;
    const float* dims[NDIMS] = {
        (const float*)d_in[0], (const float*)d_in[1], (const float*)d_in[2],
        (const float*)d_in[3], (const float*)d_in[4]};
    const int*   ei = (const int*)d_in[5];     // [5][2][E]
    const float* ew = (const float*)d_in[6];   // [5][E]
    const float* convW[6] = {(const float*)d_in[8],  (const float*)d_in[10],
                             (const float*)d_in[12], (const float*)d_in[14],
                             (const float*)d_in[16], (const float*)d_in[18]};
    const float* convb[6] = {(const float*)d_in[9],  (const float*)d_in[11],
                             (const float*)d_in[13], (const float*)d_in[15],
                             (const float*)d_in[17], (const float*)d_in[19]};
    const float* linW[6] = {(const float*)d_in[20], (const float*)d_in[22],
                            (const float*)d_in[24], (const float*)d_in[26],
                            (const float*)d_in[28], (const float*)d_in[30]};
    const float* linb[6] = {(const float*)d_in[21], (const float*)d_in[23],
                            (const float*)d_in[25], (const float*)d_in[27],
                            (const float*)d_in[29], (const float*)d_in[31]};
    const float* change1_W = (const float*)d_in[32];
    const float* change1_b = (const float*)d_in[33];
    const float* change2_W = (const float*)d_in[34];
    const float* change2_b = (const float*)d_in[35];
    const float* dimw  = (const float*)d_in[36];
    const float* ksave = (const float*)d_in[37];

    // ---- workspace carve-up ----
    const size_t NMC = (size_t)NNODES * CMAX;
    float* ws   = (float*)d_ws;
    float* gA   = ws;                            // 5*NMC  gcn f32 ping
    float* gB   = gA  + NDIMS * NMC;             // 5*NMC  gcn f32 pong
    float* oA   = gB  + NDIMS * NMC;             // 5*NMC  out ping
    float* oB   = oA  + NDIMS * NMC;             // 5*NMC  out pong
    float* deg  = oB  + NDIMS * NMC;             // 5*2048
    float* coef = deg + NDIMS * NNODES;          // 5*E
    float* H    = coef + (size_t)NDIMS * NEDGES; // NMC  x@W
    float* AG   = H   + NMC;                     // NMC  aggregated
    float* CSM  = AG  + NMC;                     // NMC  class sums
    float* CCN  = CSM + NMC;                     // 2048 class counts
    float* AV   = CCN + NNODES;                  // NMC  class avg gathered
    float* LN   = AV  + NMC;                     // NMC  lin output
    float* FIR  = LN  + NMC;                     // NMC  fir (changed skip)
    int*   FLG  = (int*)(FIR + NMC);             // 2048 class flags
    float* S    = (float*)(FLG + NNODES);        // 2048*2048 gram/laplacian
    // bf16 operand pool (all 16B aligned; strides are multiples of 16B)
    __bf16* Db   = (__bf16*)(S + (size_t)NNODES * NNODES);  // 5 * 2048*512
    __bf16* gbfA = Db   + (size_t)NDIMS * NNODES * 512;     // 5*NMC
    __bf16* gbfB = gbfA + NDIMS * NMC;                      // 5*NMC
    __bf16* XNb  = gbfB + NDIMS * NMC;                      // NMC
    __bf16* AVb  = XNb  + NMC;                              // NMC
    __bf16* OPb  = AVb  + NMC;                              // NMC
    __bf16* WT1  = OPb  + NMC;                              // 512*256 (convW^T)
    __bf16* WT2  = WT1  + 512 * 256;                        // 512*256 (linW^T)
    __bf16* WT3  = WT2  + 512 * 256;                        // 512*256 (changeW^T)

    auto cdiv = [](int a, int b) { return (a + b - 1) / b; };

    // --- one-time: bf16 input features; per-dim degree + edge coefficients ---
    for (int i = 0; i < NDIMS; ++i) {
        k_cvt<<<cdiv(NNODES * 512, 256), 256, 0, stream>>>(
            dims[i], Db + (size_t)i * NNODES * 512, NNODES * 512);
        const int* src = ei + (size_t)i * 2 * NEDGES;
        const int* dst = src + NEDGES;
        const float* ewi = ew + (size_t)i * NEDGES;
        float* degi = deg + (size_t)i * NNODES;
        float* cfi  = coef + (size_t)i * NEDGES;
        k_zero<<<cdiv(NNODES, 256), 256, 0, stream>>>(degi, NNODES);
        k_deg_scatter<<<cdiv(NEDGES, 256), 256, 0, stream>>>(dst, ewi, degi, NEDGES);
        k_deg_finish<<<cdiv(NNODES, 256), 256, 0, stream>>>(degi, NNODES);
        k_coef<<<cdiv(NEDGES, 256), 256, 0, stream>>>(src, dst, ewi, degi, cfi, NEDGES);
    }

    const __bf16* gPb[NDIMS];   // bf16 prev gcn features (GEMM A operands)
    const float*  oP[NDIMS];    // f32 prev outs (skip path)
    for (int i = 0; i < NDIMS; ++i) {
        gPb[i] = Db + (size_t)i * NNODES * 512;
        oP[i]  = dims[i];
    }

    for (int L = 0; L < 6; ++L) {
        const int Kin  = (L == 0) ? 512 : ((L == 1) ? 256 : 128);
        const int Cout = (L == 0) ? 256 : 128;
        float*  gbase  = (L & 1) ? gB : gA;
        float*  obase  = (L & 1) ? oB : oA;
        __bf16* gbbase = (L & 1) ? gbfB : gbfA;
        const __bf16* gPbn[NDIMS];
        const float*  oPn[NDIMS];

        for (int i = 0; i < NDIMS; ++i) {
            const int* src = ei + (size_t)i * 2 * NEDGES;
            const int* dst = src + NEDGES;
            const float* degi = deg + (size_t)i * NNODES;
            const float* cfi  = coef + (size_t)i * NEDGES;
            float*  gN  = gbase  + (size_t)i * NMC;
            float*  oN  = obase  + (size_t)i * NMC;
            __bf16* gNb = gbbase + (size_t)i * NMC;
            const int tot = NNODES * Cout;

            // ---- GCN: h = X@W (WMMA, async-staged), scatter-aggregate ----
            k_cvt_t<<<cdiv(Kin * Cout, 256), 256, 0, stream>>>(
                convW[L] + (size_t)i * Kin * Cout, WT1, Kin, Cout);
            {
                dim3 grd(Cout / 64, NNODES / 128);
                k_gemm_bf16<<<grd, 256, 0, stream>>>(
                    gPb[i], WT1, nullptr, H, NNODES, Kin, Cout, 0);
            }
            k_zero<<<cdiv(tot, 256), 256, 0, stream>>>(AG, tot);
            k_aggregate<<<cdiv(NEDGES * (Cout >> 2), 256), 256, 0, stream>>>(
                src, dst, cfi, H, AG, NEDGES, Cout);
            k_gcn_finish<<<cdiv(tot, 256), 256, 0, stream>>>(
                AG, H, degi, convb[L] + (size_t)i * Cout, gN, Cout);
            k_cvt<<<cdiv(tot, 256), 256, 0, stream>>>(gN, gNb, tot);

            // ---- spectral class flags: Gram (WMMA) -> Laplacian -> Jacobi ----
            k_rownorm<<<NNODES, 256, 0, stream>>>(gN, XNb, Cout);
            {
                dim3 grd(NNODES / 64, NNODES / 128);
                k_gemm_bf16<<<grd, 256, 0, stream>>>(
                    XNb, XNb, nullptr, S, NNODES, Cout, NNODES, 0);
            }
            k_laplacian<<<NNODES, 256, 0, stream>>>(S, NNODES);
            k_jacobi<<<1, 1024, 0, stream>>>(S, NNODES, JACOBI_SWEEPS);
            k_eigflags<<<1, 1024, 0, stream>>>(S, NNODES, THRESH, FLG);

            // ---- class average + lin GEMM ----
            k_zero<<<cdiv(tot, 256), 256, 0, stream>>>(CSM, tot);
            k_zero<<<cdiv(NNODES, 256), 256, 0, stream>>>(CCN, NNODES);
            k_class_sums<<<cdiv(tot, 256), 256, 0, stream>>>(FLG, gN, CSM, CCN, Cout);
            k_class_gather<<<cdiv(tot, 256), 256, 0, stream>>>(FLG, CSM, CCN, AV, Cout);
            k_cvt<<<cdiv(tot, 256), 256, 0, stream>>>(AV, AVb, tot);
            k_cvt_t<<<cdiv(Cout * Cout, 256), 256, 0, stream>>>(linW[L], WT2, Cout, Cout);
            {
                dim3 grd(Cout / 64, NNODES / 128);
                k_gemm_bf16<<<grd, 256, 0, stream>>>(
                    AVb, WT2, linb[L], LN, NNODES, Cout, Cout, 0);
            }

            // ---- skip path (change GEMMs on layers 0/1) + combine ----
            const float* firp;
            if (L == 0) {
                k_cvt_t<<<cdiv(512 * 256, 256), 256, 0, stream>>>(change1_W, WT3, 512, 256);
                dim3 grd(256 / 64, NNODES / 128);
                k_gemm_bf16<<<grd, 256, 0, stream>>>(
                    Db + (size_t)i * NNODES * 512, WT3, change1_b, FIR, NNODES, 512, 256, 0);
                firp = FIR;
            } else if (L == 1) {
                k_cvt<<<cdiv(NNODES * 256, 256), 256, 0, stream>>>(oP[i], OPb, NNODES * 256);
                k_cvt_t<<<cdiv(256 * 128, 256), 256, 0, stream>>>(change2_W, WT3, 256, 128);
                dim3 grd(128 / 64, NNODES / 128);
                k_gemm_bf16<<<grd, 256, 0, stream>>>(
                    OPb, WT3, change2_b, FIR, NNODES, 256, 128, 0);
                firp = FIR;
            } else {
                firp = oP[i];
            }
            k_combine<<<cdiv(tot, 256), 256, 0, stream>>>(firp, LN, ksave, oN, tot);

            gPbn[i] = gNb;
            oPn[i]  = oN;
        }
        for (int i = 0; i < NDIMS; ++i) { gPb[i] = gPbn[i]; oP[i] = oPn[i]; }
    }

    // ---- dimension-weighted fusion (final outs are 128 channels) ----
    const int totF = NNODES * 128;
    k_final<<<cdiv(totF, 256), 256, 0, stream>>>(
        oP[0], oP[1], oP[2], oP[3], oP[4], dimw, (float*)d_out, totF);
}
